// AlignUniformLoss_55482387529743
// MI455X (gfx1250) — matrix operand order repaired and checked by reading, not compiled, not run on hardware
//
#include <hip/hip_runtime.h>

// ---------- types ----------
typedef __bf16 v16bf __attribute__((ext_vector_type(16)));
typedef float  v8f   __attribute__((ext_vector_type(8)));
typedef unsigned int u32x4 __attribute__((ext_vector_type(4)));

union V16 { v16bf v; u32x4 q[2]; };

// ---------- constants ----------
#define BROWS   4096          // rows per input matrix
#define DIM     256           // embedding dim
#define NTOT    8192          // 2*BROWS
#define TILE    128           // gram tile edge
#define NTILES  64            // NTOT / TILE
#define LDS_PAD 8             // bf16 elems of padding per row
#define ROWB    ((DIM + LDS_PAD) * 2)   // bytes per LDS row = 528
#define DYN_LDS (2 * TILE * ROWB + 64)  // two tiles + reduction scratch

// f32 -> bf16 round-to-nearest-even (inputs are finite)
__device__ __forceinline__ unsigned short f2bf(float f) {
    unsigned u = __float_as_uint(f);
    unsigned r = u + 0x7fffu + ((u >> 16) & 1u);
    return (unsigned short)(r >> 16);
}

__device__ __forceinline__ float wave_sum(float v) {
    #pragma unroll
    for (int off = 16; off > 0; off >>= 1) v += __shfl_xor(v, off, 32);
    return v;
}

// ============================================================
// Kernel 1: per-row L2 normalize A and B, write bf16 concat x,
//           and per-row alignment partial ||a_n - b_n||^2.
// One wave per row pair. grid 512 x block 256 = 4096 waves.
// ============================================================
__global__ void au_normalize_kernel(const float* __restrict__ A,
                                    const float* __restrict__ B,
                                    unsigned short* __restrict__ xbf,
                                    float* __restrict__ alignPartial) {
    const int lane = threadIdx.x & 31;
    const int r = blockIdx.x * (blockDim.x >> 5) + (threadIdx.x >> 5);
    if (r >= BROWS) return;

    const float* arow = A + (size_t)r * DIM;
    const float* brow = B + (size_t)r * DIM;

    float av[8], bv[8];
    float sa = 0.f, sb = 0.f;
    #pragma unroll
    for (int e = 0; e < 8; ++e) {
        float xa = arow[lane + 32 * e];
        float xb = brow[lane + 32 * e];
        av[e] = xa; bv[e] = xb;
        sa += xa * xa; sb += xb * xb;
    }
    sa = wave_sum(sa);
    sb = wave_sum(sb);
    const float ia = 1.0f / fmaxf(sqrtf(sa), 1e-12f);
    const float ib = 1.0f / fmaxf(sqrtf(sb), 1e-12f);

    unsigned short* xa_out = xbf + (size_t)r * DIM;
    unsigned short* xb_out = xbf + ((size_t)r + BROWS) * DIM;

    float sd = 0.f;
    #pragma unroll
    for (int e = 0; e < 8; ++e) {
        float an = av[e] * ia;
        float bn = bv[e] * ib;
        xa_out[lane + 32 * e] = f2bf(an);
        xb_out[lane + 32 * e] = f2bf(bn);
        float d = an - bn;
        sd += d * d;
    }
    sd = wave_sum(sd);
    if (lane == 0) alignPartial[r] = sd;
}

// ============================================================
// Kernel 2: symmetric tiled Gram + exp + reduce.
// Block (ti,tj) with tj>=ti computes the 128x128 tile of
// e_ij = exp(-2 * max(2 - 2*dot(x_i,x_j), 0)) and writes its
// sum (doubled for off-diagonal tiles) to uniformPartial.
// 8 waves; wave w covers 32 I-rows x 64 J-cols (2x4 WMMA accs).
// All 6 fragments of a k-step are loaded before any WMMA so the
// DS waits are partial and the 8 WMMAs issue back-to-back.
// ============================================================
__global__ void au_gram_kernel(const unsigned short* __restrict__ xbf,
                               float* __restrict__ uniformPartial) {
    extern __shared__ char smem[];
    const int ti = blockIdx.x, tj = blockIdx.y;
    const int slot = tj * NTILES + ti;
    const int tid = threadIdx.x;

    if (tj < ti) {                     // lower triangle handled by symmetry
        if (tid == 0) uniformPartial[slot] = 0.0f;
        return;
    }

    char* ldsI = smem;
    char* ldsJ = (ti == tj) ? smem : (smem + TILE * ROWB);
    float* redbuf = (float*)(smem + 2 * TILE * ROWB);

    // ---- stage tile I (128 rows x 256 bf16) as b128 chunks ----
    {
        const u32x4* src = (const u32x4*)(xbf + (size_t)ti * TILE * DIM);
        #pragma unroll
        for (int t = 0; t < 16; ++t) {
            int c = tid + t * 256;          // 0..4095 chunk id
            int row = c >> 5, c8 = c & 31;  // 32 chunks per row
            u32x4 val = src[row * 32 + c8];
            *(u32x4*)(ldsI + row * ROWB + c8 * 16) = val;
        }
    }
    if (tj != ti) {
        const u32x4* src = (const u32x4*)(xbf + (size_t)tj * TILE * DIM);
        #pragma unroll
        for (int t = 0; t < 16; ++t) {
            int c = tid + t * 256;
            int row = c >> 5, c8 = c & 31;
            u32x4 val = src[row * 32 + c8];
            *(u32x4*)(ldsJ + row * ROWB + c8 * 16) = val;
        }
    }
    __syncthreads();

    const int wave = tid >> 5, lane = tid & 31;
    const int m  = lane & 15;       // row/col within 16-group
    const int hi = lane >> 4;       // K half select
    const int ii = (wave & 3) * 32; // wave's I row base
    const int jj = (wave >> 2) * 64;// wave's J col base

    v8f vzero = {};
    v8f acc[2][4];
    #pragma unroll
    for (int r = 0; r < 2; ++r)
        #pragma unroll
        for (int s = 0; s < 4; ++s) acc[r][s] = vzero;

    const char* aRow0 = ldsI + (size_t)(ii + m) * ROWB;
    const char* aRow1 = ldsI + (size_t)(ii + 16 + m) * ROWB;
    const char* bBase = ldsJ + (size_t)(jj + m) * ROWB;

    #pragma unroll
    for (int kb = 0; kb < DIM; kb += 32) {
        // A 16x32 bf16 layout: lanes 0-15 hold K{0..7,16..23}, lanes 16-31 K{8..15,24..31}
        const int aOff0 = (kb + hi * 8) * 2;
        const int aOff1 = (kb + 16 + hi * 8) * 2;
        // B 32x16 bf16 layout: lanes 0-15 hold K{0..7,16..23}, lanes 16-31 K{8..15,24..31} (col = lane%16)
        const int bOff = (kb + hi * 8) * 2;

        V16 a0, a1, b[4];
        a0.q[0] = *(const u32x4*)(aRow0 + aOff0);
        a0.q[1] = *(const u32x4*)(aRow0 + aOff1);
        a1.q[0] = *(const u32x4*)(aRow1 + aOff0);
        a1.q[1] = *(const u32x4*)(aRow1 + aOff1);
        #pragma unroll
        for (int s = 0; s < 4; ++s) {
            const char* bRow = bBase + (size_t)(s * 16) * ROWB + bOff;
            b[s].q[0] = *(const u32x4*)(bRow);
            b[s].q[1] = *(const u32x4*)(bRow + 32);
        }

        #pragma unroll
        for (int s = 0; s < 4; ++s) {
            acc[0][s] = __builtin_amdgcn_wmma_f32_16x16x32_bf16(
                false, a0.v, false, b[s].v, (short)0, acc[0][s], false, false);
            acc[1][s] = __builtin_amdgcn_wmma_f32_16x16x32_bf16(
                false, a1.v, false, b[s].v, (short)0, acc[1][s], false, false);
        }
    }

    // exp(-2 * max(2 - 2*dot, 0)); positions irrelevant for the sum
    float local = 0.f;
    #pragma unroll
    for (int r = 0; r < 2; ++r)
        #pragma unroll
        for (int s = 0; s < 4; ++s)
            #pragma unroll
            for (int e = 0; e < 8; ++e) {
                float dot = acc[r][s][e];
                float d2 = fmaxf(2.0f - 2.0f * dot, 0.0f);
                local += __expf(-2.0f * d2);
            }

    local = wave_sum(local);
    if (lane == 0) redbuf[wave] = local;
    __syncthreads();
    if (tid == 0) {
        float s = 0.f;
        #pragma unroll
        for (int w = 0; w < 8; ++w) s += redbuf[w];
        uniformPartial[slot] = (ti == tj) ? s : 2.0f * s;
    }
}

// ============================================================
// Kernel 3: deterministic f64 reduction + final loss assembly
// ============================================================
__global__ void au_finalize_kernel(const float* __restrict__ alignPartial,
                                   const float* __restrict__ uniformPartial,
                                   float* __restrict__ out) {
    __shared__ double sA[256];
    __shared__ double sU[256];
    const int t = threadIdx.x;
    double a = 0.0, u = 0.0;
    for (int i = t; i < BROWS; i += 256) a += (double)alignPartial[i];
    for (int i = t; i < NTILES * NTILES; i += 256) u += (double)uniformPartial[i];
    sA[t] = a; sU[t] = u;
    __syncthreads();
    for (int s = 128; s > 0; s >>= 1) {
        if (t < s) { sA[t] += sA[t + s]; sU[t] += sU[t + s]; }
        __syncthreads();
    }
    if (t == 0) {
        double align_loss = sA[0] / (double)BROWS;
        double n = (double)NTOT;
        double uniform_loss = (sU[0] - n) / (n * (n - 1.0));
        out[0] = (float)(align_loss + uniform_loss);
    }
}

// ============================================================
extern "C" void kernel_launch(void* const* d_in, const int* in_sizes, int n_in,
                              void* d_out, int out_size, void* d_ws, size_t ws_size,
                              hipStream_t stream) {
    (void)in_sizes; (void)n_in; (void)out_size; (void)ws_size;
    const float* A = (const float*)d_in[0];
    const float* B = (const float*)d_in[1];
    float* out = (float*)d_out;

    char* ws = (char*)d_ws;
    unsigned short* xbf = (unsigned short*)ws;                     // 8192*256*2 = 4 MiB
    float* alignPartial   = (float*)(ws + (size_t)NTOT * DIM * 2); // 4096 f32
    float* uniformPartial = alignPartial + BROWS;                  // 4096 f32

    au_normalize_kernel<<<dim3(BROWS / 8), dim3(256), 0, stream>>>(A, B, xbf, alignPartial);
    au_gram_kernel<<<dim3(NTILES, NTILES), dim3(256), DYN_LDS, stream>>>(xbf, uniformPartial);
    au_finalize_kernel<<<dim3(1), dim3(256), 0, stream>>>(alignPartial, uniformPartial, out);
}
